// Attention_30812095381719
// MI455X (gfx1250) — matrix operand order repaired and checked by reading, not compiled, not run on hardware
//
#include <hip/hip_runtime.h>
#include <stdint.h>

// ---------------- CDNA5 (gfx1250) wave32 WMMA attention pipeline ----------------
// bf16 WMMA everywhere + Tensor Data Mover (TDM) async tile staging.

typedef __attribute__((ext_vector_type(16))) __bf16 v16bf;
typedef __attribute__((ext_vector_type(8)))  float  v8f;
typedef __attribute__((ext_vector_type(4)))  unsigned int v4ui;
typedef __attribute__((ext_vector_type(8)))  int v8si;
typedef __attribute__((ext_vector_type(4)))  int v4si;

#define QUERY_SCALE_F 0.08838834764831845f
#define SOFT_CAP_F    50.0f
#define WINDOW        1024
#define NHEADS        32
#define NKV           8
#define HDIM          128
#define TSEQ          2048
#define DMODEL        4096
#define QKV_COLS      6144   // 4096 q | 1024 k | 1024 v

#if __has_builtin(__builtin_amdgcn_tensor_load_to_lds) && \
    __has_builtin(__builtin_amdgcn_s_wait_tensorcnt)
#define USE_TDM 1
#else
#define USE_TDM 0
#endif

__device__ __forceinline__ __bf16 f2bf(float f) {
  union { float f; unsigned u; } a; a.f = f;
  unsigned r = a.u + 0x7FFFu + ((a.u >> 16) & 1u);   // round-to-nearest-even
  union { unsigned short s; __bf16 b; } o; o.s = (unsigned short)(r >> 16);
  return o.b;
}
__device__ __forceinline__ float bf2f(__bf16 b) {
  union { unsigned short s; __bf16 b; } i; i.b = b;
  union { unsigned u; float f; } o; o.u = ((unsigned)i.s) << 16;
  return o.f;
}

union FragBF { uint4 u[2]; v16bf v; };

__device__ __forceinline__ v8f vzero8() {
  v8f z;
#pragma unroll
  for (int i = 0; i < 8; i++) z[i] = 0.0f;
  return z;
}

__device__ __forceinline__ v8f wmma_bf16(v16bf a, v16bf b, v8f c) {
  // D = A(16x32 bf16) * B(32x16 bf16) + C(16x16 f32)
  return __builtin_amdgcn_wmma_f32_16x16x32_bf16(false, a, false, b, (short)0, c,
                                                 false, false);
}

#if USE_TDM
// Issue one 2D TDM tile load: bf16 elements (data_size=2B), tile_d0 contiguous
// elems per row, tile_d1 rows with row stride `stride0` (elems). Optional LDS
// row padding (pad codes per D# group1 spec). Wave-uniform; call from one wave.
__device__ __forceinline__ void tdm_load_2d(unsigned lds_addr, const void* gptr,
                                            unsigned tensor_d0, unsigned tensor_d1,
                                            unsigned tile_d0, unsigned tile_d1,
                                            unsigned stride0, int pad_en,
                                            unsigned pad_interval, unsigned pad_amount) {
  unsigned long long ga = (unsigned long long)(uintptr_t)gptr;
  v4ui g0;
  g0[0] = 1u;                                        // count=1 valid descriptor
  g0[1] = lds_addr;                                  // LDS byte address
  g0[2] = (unsigned)(ga & 0xFFFFFFFFull);            // global_addr[31:0]
  g0[3] = (unsigned)((ga >> 32) & 0x1FFFFFFull)      // global_addr[56:32]
        | (2u << 30);                                // type = 2 ("image")
  v8si g1;
  unsigned w0 = (1u << 16);                          // data_size = 1 -> 2 bytes
  if (pad_en) w0 |= (1u << 20) | (pad_interval << 22) | (pad_amount << 25);
  g1[0] = (int)w0;
  g1[1] = (int)((tensor_d0 & 0xFFFFu) << 16);        // tensor_dim0[15:0] @ bits 63:48
  g1[2] = (int)(((tensor_d0 >> 16) & 0xFFFFu) | ((tensor_d1 & 0xFFFFu) << 16));
  g1[3] = (int)(((tensor_d1 >> 16) & 0xFFFFu) | ((tile_d0 & 0xFFFFu) << 16));
  g1[4] = (int)(tile_d1 & 0xFFFFu);                  // tile_dim1; tile_dim2 = 0
  g1[5] = (int)stride0;                              // tensor_dim0_stride[31:0]
  g1[6] = 0;                                         // stride0 hi / stride1 lo
  g1[7] = 0;
  v4si z4 = {0, 0, 0, 0};
#if defined(__clang_major__) && (__clang_major__ >= 23)
  v8si z8 = {0, 0, 0, 0, 0, 0, 0, 0};
  __builtin_amdgcn_tensor_load_to_lds(g0, g1, z4, z4, z8, 0);
#else
  __builtin_amdgcn_tensor_load_to_lds(g0, g1, z4, z4, 0);
#endif
}
#endif  // USE_TDM

// -------------------------------- fp32 -> bf16 ----------------------------------
__global__ __launch_bounds__(256) void cvt_kernel(const float* __restrict__ in,
                                                  __bf16* __restrict__ out, int n) {
  int i = blockIdx.x * blockDim.x + threadIdx.x;
  int stride = gridDim.x * blockDim.x;
  for (; i < n; i += stride) out[i] = f2bf(in[i]);
}

// -------------------------- tiled bf16 GEMM (WMMA) -------------------------------
// MODE 0: A = x_bf [2048 x 4096], B = qkv weights (blocked per head), C -> qkv bf16
// MODE 1: A = enc_bf [2048 x 4096], B = out_w bf16 [4096 x 4096] row-major, C -> f32
template <int MODE>
__global__ __launch_bounds__(256) void gemm_bf16_kernel(
    const __bf16* __restrict__ A, const __bf16* __restrict__ Bq,
    const __bf16* __restrict__ Bkv, __bf16* __restrict__ Cbf,
    float* __restrict__ Cf) {
  __shared__ __bf16 As[128][32];
  __shared__ __bf16 Bs[64][40];   // [col][k], padded to dodge bank conflicts

  const int tid = threadIdx.x;
  const int lane = tid & 31;
  const int wid = tid >> 5;
  const int wr = wid >> 1, wc = wid & 1;          // 4 x 2 wave grid
  const int m0 = blockIdx.y * 128;
  const int c0 = blockIdx.x * 64;

  const __bf16* Bbase;
  size_t ldb;
  if (MODE == 0) {
    if (c0 < 4096) {
      int n = c0 >> 7;
      Bbase = Bq + (size_t)n * DMODEL * HDIM + (c0 & 127);
    } else if (c0 < 5120) {
      int kk = (c0 - 4096) >> 7;
      Bbase = Bkv + (size_t)kk * DMODEL * HDIM + (c0 & 127);
    } else {
      int kk = (c0 - 5120) >> 7;
      Bbase = Bkv + (size_t)(NKV + kk) * DMODEL * HDIM + (c0 & 127);
    }
    ldb = HDIM;
  } else {
    Bbase = Bq + c0;
    ldb = DMODEL;
  }

  v8f acc[2][2];
#pragma unroll
  for (int i = 0; i < 2; i++)
#pragma unroll
    for (int j = 0; j < 2; j++) acc[i][j] = vzero8();

#if !USE_TDM
  const int arow = tid >> 2;         // 0..63
  const int acol = (tid & 3) * 8;    // 0,8,16,24
#endif
  const int bk = tid >> 3;           // 0..31
  const int bc = (tid & 7) * 8;      // 0..56

  const int arl = lane & 15;
  const int klo = (lane < 16) ? 0 : 8;
  const int ksb = (lane < 16) ? 0 : 8;

#if USE_TDM
  const unsigned as_lds = (unsigned)(uintptr_t)(&As[0][0]);
#endif

  for (int k0 = 0; k0 < DMODEL; k0 += 32) {
    __syncthreads();
#if USE_TDM
    // A tile 128x32 via Tensor Data Mover (one issue from wave 0)
    if (wid == 0) {
      tdm_load_2d(as_lds, &A[(size_t)m0 * DMODEL + k0],
                  /*tensor_d0=*/32, /*tensor_d1=*/128,
                  /*tile_d0=*/32, /*tile_d1=*/128,
                  /*stride0=*/DMODEL, /*pad_en=*/0, 0, 0);
    }
#else
    // A tile 128x32 (vector b128 loads/stores)
#pragma unroll
    for (int p = 0; p < 2; p++) {
      int r = p * 64 + arow;
      uint4 t = *reinterpret_cast<const uint4*>(&A[(size_t)(m0 + r) * DMODEL + k0 + acol]);
      *reinterpret_cast<uint4*>(&As[r][acol]) = t;
    }
#endif
    // B tile 32(k) x 64(col): vector global load, transpose into Bs[col][k]
    {
      uint4 t = *reinterpret_cast<const uint4*>(&Bbase[(size_t)(k0 + bk) * ldb + bc]);
      const __bf16* e = reinterpret_cast<const __bf16*>(&t);
#pragma unroll
      for (int i = 0; i < 8; i++) Bs[bc + i][bk] = e[i];
    }
#if USE_TDM
    if (wid == 0) __builtin_amdgcn_s_wait_tensorcnt((short)0);
#endif
    __syncthreads();

    v16bf af[2], bfr[2];
#pragma unroll
    for (int mi = 0; mi < 2; mi++) {
      int rr = wr * 32 + mi * 16 + arl;
      FragBF f;
      f.u[0] = *reinterpret_cast<const uint4*>(&As[rr][klo]);
      f.u[1] = *reinterpret_cast<const uint4*>(&As[rr][klo + 16]);
      af[mi] = f.v;
    }
#pragma unroll
    for (int ni = 0; ni < 2; ni++) {
      int cc = wc * 32 + ni * 16 + arl;
      FragBF f;
      f.u[0] = *reinterpret_cast<const uint4*>(&Bs[cc][ksb]);
      f.u[1] = *reinterpret_cast<const uint4*>(&Bs[cc][ksb + 16]);
      bfr[ni] = f.v;
    }
#pragma unroll
    for (int mi = 0; mi < 2; mi++)
#pragma unroll
      for (int ni = 0; ni < 2; ni++)
        acc[mi][ni] = wmma_bf16(af[mi], bfr[ni], acc[mi][ni]);
  }

  const int rowofs = (lane < 16) ? 0 : 8;
#pragma unroll
  for (int mi = 0; mi < 2; mi++)
#pragma unroll
    for (int ni = 0; ni < 2; ni++)
#pragma unroll
      for (int r = 0; r < 8; r++) {
        int row = m0 + wr * 32 + mi * 16 + rowofs + r;
        int col = c0 + wc * 32 + ni * 16 + arl;
        if (MODE == 0)
          Cbf[(size_t)row * QKV_COLS + col] = f2bf(acc[mi][ni][r]);
        else
          Cf[(size_t)row * DMODEL + col] = acc[mi][ni][r];
      }
}

// ------------------------------- RoPE (+q scale) ---------------------------------
__global__ __launch_bounds__(256) void rope_kernel(__bf16* __restrict__ qkv,
                                                   const int* __restrict__ seg) {
  int idx = blockIdx.x * blockDim.x + threadIdx.x;
  const int total = TSEQ * (NHEADS + NKV) * (HDIM / 2);
  if (idx >= total) return;
  int hp = idx & 63;
  int rem = idx >> 6;
  int slot = rem % (NHEADS + NKV);
  int t = rem / (NHEADS + NKV);
  size_t base;
  float scale;
  if (slot < NHEADS) {
    base = (size_t)t * QKV_COLS + slot * HDIM;
    scale = QUERY_SCALE_F;
  } else {
    base = (size_t)t * QKV_COLS + 4096 + (slot - NHEADS) * HDIM;
    scale = 1.0f;
  }
  float pos = (float)seg[t];
  float frac = (float)hp * (1.0f / 64.0f);
  float inv_ts = expf(-frac * 9.210340371976184f);  // 10000^(-frac)
  float ang = pos * inv_ts;
  float s = sinf(ang), c = cosf(ang);
  float x1 = bf2f(qkv[base + hp]);
  float x2 = bf2f(qkv[base + 64 + hp]);
  qkv[base + hp] = f2bf((x1 * c - x2 * s) * scale);
  qkv[base + 64 + hp] = f2bf((x2 * c + x1 * s) * scale);
}

// ------------------- flash attention (softcap + window + causal) -----------------
__global__ __launch_bounds__(256) void attn_kernel(const __bf16* __restrict__ qkv,
                                                   __bf16* __restrict__ enc) {
  __shared__ __bf16 Ks[64][136];    // [key][h], padded (TDM pad: 4 dwords / 64 dwords)
  __shared__ __bf16 Vst[128][72];   // [h][key], padded
  __shared__ __bf16 Pst[8][16][64]; // per-wave probs, A-layout staging

  const int tid = threadIdx.x;
  const int lane = tid & 31, wid = tid >> 5;
  const int head = blockIdx.y;
  const int kvh = head >> 2;                // G = NHEADS / NKV = 4
  const int t0 = blockIdx.x * 128;
  const int tq0 = t0 + wid * 16;

  const int arl = lane & 15;
  const int klo = (lane < 16) ? 0 : 8;
  const int khalf = (lane < 16) ? 0 : 8;
  const int rowofs = (lane < 16) ? 0 : 8;

  // preload Q fragments (16 rows x 128, 4 k-steps of 32), A-layout
  v16bf qf[4];
  {
    const __bf16* qrow = qkv + (size_t)(tq0 + arl) * QKV_COLS + head * HDIM;
#pragma unroll
    for (int hs = 0; hs < 4; hs++) {
      FragBF f;
      f.u[0] = *reinterpret_cast<const uint4*>(&qrow[hs * 32 + klo]);
      f.u[1] = *reinterpret_cast<const uint4*>(&qrow[hs * 32 + klo + 16]);
      qf[hs] = f.v;
    }
  }

  v8f o[8];
#pragma unroll
  for (int i = 0; i < 8; i++) o[i] = vzero8();
  float m[8], l[8];
#pragma unroll
  for (int r = 0; r < 8; r++) { m[r] = -1e30f; l[r] = 0.0f; }

  int lo = t0 - (WINDOW - 1);
  if (lo < 0) lo = 0;
  const int sbeg = lo & ~63;
  const int send = (t0 + 128 < TSEQ) ? (t0 + 128) : TSEQ;

  const int ldrow = tid >> 2;        // key 0..63
  const int ldcol = (tid & 3) * 32;  // h 0,32,64,96

#if USE_TDM
  const unsigned ks_lds = (unsigned)(uintptr_t)(&Ks[0][0]);
#endif

  for (int s0 = sbeg; s0 < send; s0 += 64) {
    __syncthreads();
#if USE_TDM
    // K tile 64x128 via TDM with hardware LDS row padding -> Ks[64][136]
    if (wid == 0) {
      tdm_load_2d(ks_lds,
                  qkv + (size_t)s0 * QKV_COLS + 4096 + kvh * HDIM,
                  /*tensor_d0=*/HDIM, /*tensor_d1=*/TSEQ,
                  /*tile_d0=*/HDIM, /*tile_d1=*/64,
                  /*stride0=*/QKV_COLS,
                  /*pad_en=*/1, /*pad_interval=*/5 /* every 64 dwords */,
                  /*pad_amount=*/3 /* 4 dwords */);
    }
#endif
    // stage V transposed [h][s] (and K manually when no TDM)
    {
      const __bf16* vrow = qkv + (size_t)(s0 + ldrow) * QKV_COLS + 5120 + kvh * HDIM + ldcol;
#if !USE_TDM
      const __bf16* krow = qkv + (size_t)(s0 + ldrow) * QKV_COLS + 4096 + kvh * HDIM + ldcol;
#endif
#pragma unroll
      for (int i = 0; i < 4; i++) {
#if !USE_TDM
        uint4 tk = *reinterpret_cast<const uint4*>(&krow[i * 8]);
        *reinterpret_cast<uint4*>(&Ks[ldrow][ldcol + i * 8]) = tk;
#endif
        uint4 tv = *reinterpret_cast<const uint4*>(&vrow[i * 8]);
        const __bf16* e = reinterpret_cast<const __bf16*>(&tv);
#pragma unroll
        for (int j = 0; j < 8; j++) Vst[ldcol + i * 8 + j][ldrow] = e[j];
      }
    }
#if USE_TDM
    if (wid == 0) __builtin_amdgcn_s_wait_tensorcnt((short)0);
#endif
    __syncthreads();

    // logits: 4 key sub-tiles of 16, K-dim = H = 128 (4 WMMA steps)
    v8f st[4];
#pragma unroll
    for (int ct = 0; ct < 4; ct++) {
      v8f s = vzero8();
      int krow16 = ct * 16 + arl;
#pragma unroll
      for (int hs = 0; hs < 4; hs++) {
        FragBF f;
        f.u[0] = *reinterpret_cast<const uint4*>(&Ks[krow16][hs * 32 + khalf]);
        f.u[1] = *reinterpret_cast<const uint4*>(&Ks[krow16][hs * 32 + khalf + 16]);
        s = wmma_bf16(qf[hs], f.v, s);
      }
      st[ct] = s;
    }

    // softcap + mask + online softmax (per-row stats replicated across 16 lanes)
    float a_r[8];
#pragma unroll
    for (int r = 0; r < 8; r++) {
      int t = tq0 + r + rowofs;
      float vmax = -1e30f;
      float vv[4];
#pragma unroll
      for (int ct = 0; ct < 4; ct++) {
        int key = s0 + ct * 16 + arl;
        float v = st[ct][r];
        v = tanhf(v * (1.0f / SOFT_CAP_F)) * SOFT_CAP_F;
        bool ok = (key <= t) && (key > t - WINDOW);
        v = ok ? v : -1e30f;
        vv[ct] = v;
        vmax = fmaxf(vmax, v);
      }
#pragma unroll
      for (int ml = 1; ml < 16; ml <<= 1) vmax = fmaxf(vmax, __shfl_xor(vmax, ml, 32));
      float mnew = fmaxf(m[r], vmax);
      float alpha = expf(m[r] - mnew);
      m[r] = mnew;
      float ls = 0.0f;
#pragma unroll
      for (int ct = 0; ct < 4; ct++) {
        float p = (vv[ct] <= -1e29f) ? 0.0f : expf(vv[ct] - mnew);
        ls += p;
        Pst[wid][r + rowofs][ct * 16 + arl] = f2bf(p);
      }
#pragma unroll
      for (int ml = 1; ml < 16; ml <<= 1) ls += __shfl_xor(ls, ml, 32);
      l[r] = l[r] * alpha + ls;
      a_r[r] = alpha;
    }
    // rescale accumulators
#pragma unroll
    for (int ht = 0; ht < 8; ht++)
#pragma unroll
      for (int r = 0; r < 8; r++) o[ht][r] *= a_r[r];

    __syncthreads();  // Pst visible (uniform barrier across block)

    // O += P(16x64) * V(64x128)
    v16bf pf[2];
#pragma unroll
    for (int kb = 0; kb < 2; kb++) {
      FragBF f;
      f.u[0] = *reinterpret_cast<const uint4*>(&Pst[wid][arl][kb * 32 + klo]);
      f.u[1] = *reinterpret_cast<const uint4*>(&Pst[wid][arl][kb * 32 + klo + 16]);
      pf[kb] = f.v;
    }
#pragma unroll
    for (int ht = 0; ht < 8; ht++) {
      v8f acc = o[ht];
#pragma unroll
      for (int kb = 0; kb < 2; kb++) {
        FragBF f;
        f.u[0] = *reinterpret_cast<const uint4*>(&Vst[ht * 16 + arl][kb * 32 + khalf]);
        f.u[1] = *reinterpret_cast<const uint4*>(&Vst[ht * 16 + arl][kb * 32 + khalf + 16]);
        acc = wmma_bf16(pf[kb], f.v, acc);
      }
      o[ht] = acc;
    }
  }

  // normalize + store encoded [t][n*128+h] as bf16
#pragma unroll
  for (int r = 0; r < 8; r++) {
    float inv = (l[r] > 0.0f) ? (1.0f / l[r]) : 0.0f;
    int t = tq0 + r + rowofs;
    size_t base = (size_t)t * DMODEL + head * HDIM;
#pragma unroll
    for (int ht = 0; ht < 8; ht++) enc[base + ht * 16 + arl] = f2bf(o[ht][r] * inv);
  }
}

// ----------------------------------- launch --------------------------------------
extern "C" void kernel_launch(void* const* d_in, const int* in_sizes, int n_in,
                              void* d_out, int out_size, void* d_ws, size_t ws_size,
                              hipStream_t stream) {
  (void)in_sizes; (void)n_in; (void)out_size; (void)ws_size;
  const float* x = (const float*)d_in[0];
  const int* seg = (const int*)d_in[1];
  // d_in[2] = attn_mask (tril) — implied analytically by the causal mask
  const float* q_w = (const float*)d_in[3];
  const float* kv_w = (const float*)d_in[4];
  const float* out_w = (const float*)d_in[5];
  float* out = (float*)d_out;

  char* ws = (char*)d_ws;
  const size_t SZ_X = (size_t)TSEQ * DMODEL * 2;
  const size_t SZ_QW = (size_t)NHEADS * DMODEL * HDIM * 2;
  const size_t SZ_KVW = (size_t)2 * NKV * DMODEL * HDIM * 2;
  const size_t SZ_OW = (size_t)NHEADS * HDIM * DMODEL * 2;
  const size_t SZ_QKV = (size_t)TSEQ * QKV_COLS * 2;
  __bf16* x_bf = (__bf16*)(ws);
  __bf16* qw_bf = (__bf16*)(ws + SZ_X);
  __bf16* kvw_bf = (__bf16*)(ws + SZ_X + SZ_QW);
  __bf16* ow_bf = (__bf16*)(ws + SZ_X + SZ_QW + SZ_KVW);
  __bf16* qkv_bf = (__bf16*)(ws + SZ_X + SZ_QW + SZ_KVW + SZ_OW);
  __bf16* enc_bf = (__bf16*)(ws + SZ_X + SZ_QW + SZ_KVW + SZ_OW + SZ_QKV);

  cvt_kernel<<<2048, 256, 0, stream>>>(x, x_bf, TSEQ * DMODEL);
  cvt_kernel<<<2048, 256, 0, stream>>>(q_w, qw_bf, NHEADS * DMODEL * HDIM);
  cvt_kernel<<<2048, 256, 0, stream>>>(kv_w, kvw_bf, 2 * NKV * DMODEL * HDIM);
  cvt_kernel<<<2048, 256, 0, stream>>>(out_w, ow_bf, NHEADS * HDIM * DMODEL);

  gemm_bf16_kernel<0><<<dim3(QKV_COLS / 64, TSEQ / 128), 256, 0, stream>>>(
      x_bf, qw_bf, kvw_bf, qkv_bf, nullptr);
  rope_kernel<<<(TSEQ * (NHEADS + NKV) * (HDIM / 2) + 255) / 256, 256, 0, stream>>>(
      qkv_bf, seg);
  attn_kernel<<<dim3(TSEQ / 128, NHEADS), 256, 0, stream>>>(qkv_bf, enc_bf);
  gemm_bf16_kernel<1><<<dim3(DMODEL / 64, TSEQ / 128), 256, 0, stream>>>(
      enc_bf, ow_bf, nullptr, nullptr, out);
}
